// Centroids_32057635897630
// MI455X (gfx1250) — matrix operand order repaired and checked by reading, not compiled, not run on hardware
//
#include <hip/hip_runtime.h>
#include <stdint.h>

#define N_TOK  32768
#define N_FEAT 512
#define N_CENT 8192
#define DECAYF 0.99f
#define ONEMF  0.01f
#define EPSF   1e-5f

typedef __attribute__((ext_vector_type(16))) __bf16         v16bf;
typedef __attribute__((ext_vector_type(8)))  float          v8f;
typedef __attribute__((ext_vector_type(16))) unsigned short v16us;
typedef __attribute__((ext_vector_type(8)))  unsigned short v8us;

__device__ __forceinline__ unsigned short f2bf(float f) {
  union { float f; unsigned u; } c; c.f = f;
  unsigned u = c.u;
  u += 0x7FFFu + ((u >> 16) & 1u);   // round-to-nearest-even
  return (unsigned short)(u >> 16);
}

// ---------------- init: zero atomic-accumulation regions + loss slot ----------------
__global__ void k_init(float* stats, int nzero, float* lossSlot) {
  int i = blockIdx.x * blockDim.x + threadIdx.x;
  if (i < nzero) stats[i] = 0.f;
  if (i == 0) *lossSlot = 0.f;
}

// ---------------- x (f32) -> xb (bf16), elementwise ----------------
__global__ void k_xb(const float* __restrict__ x, unsigned short* __restrict__ xb, int n) {
  int i = blockIdx.x * blockDim.x + threadIdx.x;
  if (i < n) xb[i] = f2bf(x[i]);
}

// ---------------- C[d][k] (f32) -> ct[k][d] (bf16), LDS 32x32 tile transpose ----------------
__global__ void k_ct(const float* __restrict__ C, unsigned short* __restrict__ ct) {
  __shared__ float t[32][33];
  int k0 = blockIdx.x * 32;
  int d0 = blockIdx.y * 32;
  int tx = threadIdx.x & 31;
  int ty = threadIdx.x >> 5;           // 0..7
  #pragma unroll
  for (int i = 0; i < 4; ++i) {
    int d = d0 + ty + 8 * i;
    t[ty + 8 * i][tx] = C[(size_t)d * N_CENT + (k0 + tx)];
  }
  __syncthreads();
  #pragma unroll
  for (int i = 0; i < 4; ++i) {
    int k = k0 + ty + 8 * i;
    ct[(size_t)k * N_FEAT + (d0 + tx)] = f2bf(t[tx][ty + 8 * i]);
  }
}

// ---------------- cnorm[k] = sum_d C[d][k]^2 ----------------
__global__ void k_cnorm(const float* __restrict__ C, float* __restrict__ cnorm) {
  int k = blockIdx.x * blockDim.x + threadIdx.x;
  float s = 0.f;
  for (int d = 0; d < N_FEAT; ++d) {
    float c = C[(size_t)d * N_CENT + k];
    s += c * c;
  }
  cnorm[k] = s;
}

// ============ assign: bf16 WMMA distance GEMM, async-to-LDS triple-buffered B ============
// Block = 256 threads = 8 waves; each wave owns 16 token rows (128 rows/block).
// B tile (16 centroids x 512 features, bf16, 16 KB) lives in LDS pre-swizzled as
// [operand rb][lane][32B]; compute reads are software-pipelined one operand ahead so the
// backend can use partial s_wait_dscnt. Tiles stream in two iterations ahead via
// global_load_async_to_lds_b128; async loads complete in order, so s_wait_asynccnt <= 4
// (one tile's worth per wave) guarantees tile kt+1 resident while kt+2 stays in flight.
__global__ __launch_bounds__(256, 1)
void k_assign(const unsigned short* __restrict__ xb,
              const unsigned short* __restrict__ ct,
              const float* __restrict__ cnorm,
              int* __restrict__ ind) {
  constexpr int NT      = N_CENT / 16;   // 512 tiles
  constexpr int TILE_US = 16 * 512;      // ushorts per tile buffer (16 KB)

  __shared__ unsigned short sb[3 * TILE_US];   // triple buffer, 48 KB

  const int tid   = threadIdx.x;
  const int lane  = tid & 31;
  const int wave  = tid >> 5;            // 0..7
  const int mbase = (blockIdx.x * 8 + wave) * 16;
  const int half  = lane >> 4;           // 0 or 1
  const int l15   = lane & 15;

  // ---- async-staging setup: 1024 subchunks of 16B, 4 per thread ----
  // subchunk c: s=c&1 (16B half), l=(c>>1)&31 (dest lane), rb=c>>6 (operand)
  unsigned long long ga[4];   // global source addresses for tile 0
  unsigned int       lo[4];   // LDS dest byte addresses, buffer 0
  #pragma unroll
  for (int j = 0; j < 4; ++j) {
    const int c  = tid + 256 * j;
    const int s  = c & 1;
    const int l  = (c >> 1) & 31;
    const int rb = c >> 6;
    const int uidx = rb * 512 + l * 16 + s * 8;                        // ushort off in tile buf
    const int gidx = (l & 15) * 512 + rb * 32 + (l >> 4) * 8 + s * 16; // ushort off in ct tile
    lo[j] = (unsigned int)(size_t)(&sb[uidx]);
    ga[j] = (unsigned long long)(size_t)(ct + gidx);
  }

  auto issueTile = [&](int kt, int bufsel) {
    const unsigned int boff = (unsigned int)(bufsel * (TILE_US * 2));  // bytes
    #pragma unroll
    for (int j = 0; j < 4; ++j) {
      unsigned long long g = ga[j] + (unsigned long long)kt * (16u * 512u * 2u);
      unsigned int       d = lo[j] + boff;
      asm volatile("global_load_async_to_lds_b128 %0, %1, off"
                   :: "v"(d), "v"(g) : "memory");
    }
  };

  // ---- A operands: 16 tiles of 16x32 bf16, resident in VGPRs for whole kernel ----
  const unsigned short* xrow = xb + (size_t)(mbase + l15) * N_FEAT;
  const int kb = half * 8;
  v16bf A[16];
  #pragma unroll
  for (int rb = 0; rb < 16; ++rb) {
    const unsigned short* p = xrow + rb * 32 + kb;
    v8us plo = *(const v8us*)(p);        // K = rb*32+kb .. +7
    v8us phi = *(const v8us*)(p + 16);   // K = rb*32+kb+16 .. +23
    v16us w = __builtin_shufflevector(plo, phi, 0,1,2,3,4,5,6,7,8,9,10,11,12,13,14,15);
    A[rb] = __builtin_bit_cast(v16bf, w);
  }

  float bestV[8];
  int   bestI[8];
  #pragma unroll
  for (int r = 0; r < 8; ++r) { bestV[r] = 3.4e38f; bestI[r] = 0; }

  // ---- prologue: tiles 0 and 1 in flight; async loads retire in order ----
  issueTile(0, 0);
  issueTile(1, 1);
  asm volatile("s_wait_asynccnt 0x4" ::: "memory");   // tile 0 resident
  __syncthreads();

  int cb = 0;                                         // compute buffer for tile kt
  #pragma unroll 1
  for (int kt = 0; kt < NT; ++kt) {
    // ---- stream tile kt+2 two buffers ahead ----
    if (kt + 2 < NT) {
      int pb = cb + 2; if (pb >= 3) pb -= 3;
      issueTile(kt + 2, pb);
    }

    const int curBase = cb * TILE_US;
    const float cn = cnorm[kt * 16 + l15];            // hoisted: overlaps WMMA chain

    // ---- 16 chained WMMAs, DS reads pipelined one operand ahead ----
    v16us bw = *(const v16us*)(&sb[curBase + lane * 16]);
    v8f acc = {0.f, 0.f, 0.f, 0.f, 0.f, 0.f, 0.f, 0.f};
    #pragma unroll
    for (int rb = 0; rb < 16; ++rb) {
      v16us bwn;
      if (rb < 15) bwn = *(const v16us*)(&sb[curBase + (rb + 1) * 512 + lane * 16]);
      v16bf B = __builtin_bit_cast(v16bf, bw);
      acc = __builtin_amdgcn_wmma_f32_16x16x32_bf16(false, A[rb], false, B,
                                                    (short)0, acc, false, false);
      if (rb < 15) bw = bwn;
    }

    // ---- running argmin (||x||^2 dropped: constant per row) ----
    const int n = kt * 16 + l15;
    #pragma unroll
    for (int r = 0; r < 8; ++r) {
      float d = cn - 2.0f * acc[r];
      if (d < bestV[r]) { bestV[r] = d; bestI[r] = n; }
    }

    // tile kt+1 (older async group) retired; kt+2 may remain in flight
    asm volatile("s_wait_asynccnt 0x4" ::: "memory");
    __syncthreads();

    cb = (cb == 2) ? 0 : cb + 1;
  }

  // cross-lane argmin within each 16-lane half (ties -> smaller index)
  #pragma unroll
  for (int m = 8; m >= 1; m >>= 1) {
    #pragma unroll
    for (int r = 0; r < 8; ++r) {
      float ov = __shfl_xor(bestV[r], m, 32);
      int   oi = __shfl_xor(bestI[r], m, 32);
      if (ov < bestV[r] || (ov == bestV[r] && oi < bestI[r])) {
        bestV[r] = ov; bestI[r] = oi;
      }
    }
  }
  // D layout: lane<16 VGPR r -> row r; lane>=16 VGPR r -> row 8+r
  if (l15 == 0) {
    #pragma unroll
    for (int r = 0; r < 8; ++r)
      ind[mbase + half * 8 + r] = bestI[r];
  }
}

// ---------------- scatter stats: counts + per-cluster x sums ----------------
__global__ void k_stats(const float* __restrict__ x, const int* __restrict__ ind,
                        float* __restrict__ counts, float* __restrict__ centSum) {
  const int t = blockIdx.x;
  const int k = ind[t];
  const float* xr = x + (size_t)t * N_FEAT;
  float* cs = centSum + (size_t)k * N_FEAT;
  for (int d = threadIdx.x; d < N_FEAT; d += blockDim.x)
    atomicAdd(&cs[d], xr[d]);
  if (threadIdx.x == 0) atomicAdd(&counts[k], 1.0f);
}

// ---------------- EMA cluster sizes + global sum n ----------------
__global__ void k_ema(const float* __restrict__ clusterSize, const float* __restrict__ counts,
                      float* __restrict__ newCS, float* __restrict__ nsum) {
  __shared__ float red[256];
  int k = blockIdx.x * blockDim.x + threadIdx.x;
  float v = DECAYF * clusterSize[k] + ONEMF * counts[k];
  newCS[k] = v;
  red[threadIdx.x] = v;
  __syncthreads();
  for (int s = 128; s > 0; s >>= 1) {
    if (threadIdx.x < s) red[threadIdx.x] += red[threadIdx.x + s];
    __syncthreads();
  }
  if (threadIdx.x == 0) atomicAdd(nsum, red[0]);
}

// ---------------- Laplace-smoothed inverse cluster size ----------------
__global__ void k_invcs(const float* __restrict__ newCS, const float* __restrict__ nsum,
                        float* __restrict__ invcs) {
  int k = blockIdx.x * blockDim.x + threadIdx.x;
  float n  = *nsum;
  float cs = (newCS[k] + EPSF) / (n + (float)N_CENT * EPSF) * n;
  invcs[k] = 1.0f / cs;
}

// ---------------- gather updated centroids, emit x_quant + loss ----------------
__global__ void k_quant(const float* __restrict__ x, const int* __restrict__ ind,
                        const float* __restrict__ avg, const float* __restrict__ centSum,
                        const float* __restrict__ invcs,
                        float* __restrict__ out, float* __restrict__ lossSlot) {
  __shared__ float red[256];
  const int t = blockIdx.x;
  const int k = ind[t];
  const float ics = invcs[k];
  float ls = 0.f;
  for (int d = threadIdx.x; d < N_FEAT; d += blockDim.x) {
    float q = (DECAYF * avg[(size_t)d * N_CENT + k] +
               ONEMF  * centSum[(size_t)k * N_FEAT + d]) * ics;
    out[(size_t)t * N_FEAT + d] = q;            // x_quant == quantize numerically
    float diff = q - x[(size_t)t * N_FEAT + d];
    ls += diff * diff;
  }
  red[threadIdx.x] = ls;
  __syncthreads();
  for (int s = 128; s > 0; s >>= 1) {
    if (threadIdx.x < s) red[threadIdx.x] += red[threadIdx.x + s];
    __syncthreads();
  }
  if (threadIdx.x == 0)
    atomicAdd(lossSlot, red[0] * (1.0f / ((float)N_TOK * (float)N_FEAT)));
}

extern "C" void kernel_launch(void* const* d_in, const int* in_sizes, int n_in,
                              void* d_out, int out_size, void* d_ws, size_t ws_size,
                              hipStream_t stream) {
  const float* x   = (const float*)d_in[0];   // [32768, 512]
  const float* C   = (const float*)d_in[1];   // [512, 8192]
  const float* csz = (const float*)d_in[2];   // [8192]
  const float* avg = (const float*)d_in[3];   // [512, 8192]
  float* out      = (float*)d_out;
  float* lossSlot = out + (size_t)N_TOK * N_FEAT;

  // workspace layout
  char* w = (char*)d_ws;
  unsigned short* xb = (unsigned short*)w;  w += (size_t)N_TOK  * N_FEAT * 2;  // 32 MB
  unsigned short* ct = (unsigned short*)w;  w += (size_t)N_CENT * N_FEAT * 2;  //  8 MB
  float* cnorm  = (float*)w;                w += (size_t)N_CENT * 4;
  int*   ind    = (int*)w;                  w += (size_t)N_TOK  * 4;
  float* counts = (float*)w;                w += (size_t)N_CENT * 4;           // zeroed
  float* centSum= (float*)w;                w += (size_t)N_CENT * N_FEAT * 4;  // zeroed
  float* nsum   = (float*)w;                w += 4;                            // zeroed
  float* newCS  = (float*)w;                w += (size_t)N_CENT * 4;
  float* invcs  = (float*)w;                w += (size_t)N_CENT * 4;

  const int nzero = N_CENT + N_CENT * N_FEAT + 1;  // counts + centSum + nsum (contiguous)
  k_init <<<(nzero + 255) / 256, 256, 0, stream>>>(counts, nzero, lossSlot);
  k_xb   <<<(N_TOK * N_FEAT + 255) / 256, 256, 0, stream>>>(x, xb, N_TOK * N_FEAT);
  dim3 gct(N_CENT / 32, N_FEAT / 32);
  k_ct   <<<gct, 256, 0, stream>>>(C, ct);
  k_cnorm<<<N_CENT / 256, 256, 0, stream>>>(C, cnorm);
  k_assign<<<N_TOK / 128, 256, 0, stream>>>(xb, ct, cnorm, ind);
  k_stats<<<N_TOK, 256, 0, stream>>>(x, ind, counts, centSum);
  k_ema  <<<N_CENT / 256, 256, 0, stream>>>(csz, counts, newCS, nsum);
  k_invcs<<<N_CENT / 256, 256, 0, stream>>>(newCS, nsum, invcs);
  k_quant<<<N_TOK, 256, 0, stream>>>(x, ind, avg, centSum, invcs, out, lossSlot);
}
